// Attention_42494406426794
// MI455X (gfx1250) — matrix operand order repaired and checked by reading, not compiled, not run on hardware
//
#include <hip/hip_runtime.h>
#include <cstdint>

// ---------------------------------------------------------------------------
// CDNA5 / gfx1250 attention block (GPT-2 style), bf16 WMMA + f32 accumulate.
// Dense GEMMs: 32x64 register tile per wave (2 A-subtiles x 4 B-frags = 8
// WMMAs per k-step -> halves B-operand traffic), 32-row A tile staged to LDS
// via gfx1250 async-to-LDS, fully-unrolled K=1024.
// B=4, S=2048, D=1024, H=16, hd=64
// ---------------------------------------------------------------------------

typedef __attribute__((ext_vector_type(16))) __bf16 v16bf;
typedef __attribute__((ext_vector_type(8)))  __bf16 bf16x8;
typedef __attribute__((ext_vector_type(8)))  float  v8f;

#define BATCH  4
#define SEQ    2048
#define DMODEL 1024
#define NHEAD  16
#define HD     64
#define BS     (BATCH * SEQ)      // 8192 rows
#define LDSA_STRIDE 1032          // 1024 + 8 bf16 pad -> keeps 16B align, skews banks

// -------------------------- fragment load helpers ---------------------------
// A-fragment (16x32 bf16, MxK), ISA 7.12.2: lane=(m=lane&15, h=lane>>4),
// VGPR v0..3 hold K=8h+2v(+1), v4..7 hold K=16+8h+2(v-4)(+1).
// p must point at rowptr[m] + kb + 8*h ; chunks are p[0..7] and p[16..23].
__device__ __forceinline__ v16bf load_frag_a(const __bf16* p) {
  bf16x8 lo = *(const bf16x8*)(p);
  bf16x8 hi = *(const bf16x8*)(p + 16);
  v16bf a;
#pragma unroll
  for (int i = 0; i < 8; ++i) { a[i] = lo[i]; a[8 + i] = hi[i]; }
  return a;
}

// B-fragment (32x16 bf16, KxN) from a K-contiguous (transposed) layout:
// lane=(n=lane&15, h), same K order as A: v0..3 K=8h+2v(+1), v4..7 K=16+8h+2(v-4)(+1).
// p must point at colptr[n] + kb + 8*h ; chunks are p[0..7] and p[16..23].
__device__ __forceinline__ v16bf load_frag_b(const __bf16* p) {
  return load_frag_a(p);
}

// ------------------------------ convert kernels -----------------------------
__global__ __launch_bounds__(256) void cvt_f32_bf16(const float* __restrict__ in,
                                                    __bf16* __restrict__ out, int n) {
  int i = blockIdx.x * 256 + threadIdx.x;
  if (i < n) out[i] = (__bf16)in[i];
}

// in: [K][N] fp32 row-major  ->  out: [N][K] bf16 (transposed)
__global__ __launch_bounds__(256) void tcvt_f32_bf16(const float* __restrict__ in,
                                                     __bf16* __restrict__ out,
                                                     int K, int N) {
  int i = blockIdx.x * 256 + threadIdx.x;
  if (i < K * N) {
    int k = i / N, n = i - k * N;
    out[(size_t)n * K + k] = (__bf16)in[i];
  }
}

// --------------------- async stage of 32x1024 bf16 A tile -------------------
// The tile is a contiguous 64 KB slab of the activation buffer; scatter it to
// row-padded LDS with the gfx1250 async-to-LDS path (ASYNCcnt tracked).
__device__ __forceinline__ void stage_a_tile32(const __bf16* gsrc,
                                               __bf16* ldsA, int tid) {
  const uint32_t lbase = (uint32_t)(uintptr_t)ldsA;       // LDS byte offset
  const uint64_t gbase = (uint64_t)(uintptr_t)gsrc;
#pragma unroll
  for (int i = 0; i < 16; ++i) {
    const int idx = i * 256 + tid;        // 4096 chunks of 16B
    const int row = idx >> 7;             // 128 chunks per 2048B row
    const int off = (idx & 127) << 4;
    const uint32_t laddr = lbase + (uint32_t)(row * (LDSA_STRIDE * 2) + off);
    const uint64_t gaddr = gbase + (uint64_t)(row * 2048 + off);
    asm volatile("global_load_async_to_lds_b128 %0, %1, off"
                 :: "v"(laddr), "v"(gaddr) : "memory");
  }
  asm volatile("s_wait_asynccnt 0" ::: "memory");
}

// --------------- dense mainloop: 32 rows x 64 cols per wave -----------------
// ap0/ap1: LDS A row bases (rows m / m+16), bp: B col base (+m*D + 8h).
// 2 A-frags x 4 B-frags -> 8 WMMAs per k-step (24 B/lane operand per WMMA).
// Fully unrolled K=1024: straight-line, no loop-carried register copies.
__device__ __forceinline__ void mm_k1024_2x4(v8f acc[8], const __bf16* ap0,
                                             const __bf16* ap1,
                                             const __bf16* bp) {
#pragma unroll
  for (int kb = 0; kb < DMODEL; kb += 32) {
    v16bf a0 = load_frag_a(ap0 + kb);
    v16bf a1 = load_frag_a(ap1 + kb);
    v16bf b0 = load_frag_b(bp + 0 * 16 * DMODEL + kb);
    v16bf b1 = load_frag_b(bp + 1 * 16 * DMODEL + kb);
    v16bf b2 = load_frag_b(bp + 2 * 16 * DMODEL + kb);
    v16bf b3 = load_frag_b(bp + 3 * 16 * DMODEL + kb);
    acc[0] = __builtin_amdgcn_wmma_f32_16x16x32_bf16(false, a0, false, b0,
                                                     (short)0, acc[0], false, false);
    acc[1] = __builtin_amdgcn_wmma_f32_16x16x32_bf16(false, a0, false, b1,
                                                     (short)0, acc[1], false, false);
    acc[2] = __builtin_amdgcn_wmma_f32_16x16x32_bf16(false, a0, false, b2,
                                                     (short)0, acc[2], false, false);
    acc[3] = __builtin_amdgcn_wmma_f32_16x16x32_bf16(false, a0, false, b3,
                                                     (short)0, acc[3], false, false);
    acc[4] = __builtin_amdgcn_wmma_f32_16x16x32_bf16(false, a1, false, b0,
                                                     (short)0, acc[4], false, false);
    acc[5] = __builtin_amdgcn_wmma_f32_16x16x32_bf16(false, a1, false, b1,
                                                     (short)0, acc[5], false, false);
    acc[6] = __builtin_amdgcn_wmma_f32_16x16x32_bf16(false, a1, false, b2,
                                                     (short)0, acc[6], false, false);
    acc[7] = __builtin_amdgcn_wmma_f32_16x16x32_bf16(false, a1, false, b3,
                                                     (short)0, acc[7], false, false);
  }
}

// ------------------------------- QKV GEMM -----------------------------------
// Block: 32 rows x 512 cols (8 waves x 64 cols); A tile shared via LDS.
// xb:[BS][D] bf16, wt:[3D][D] bf16 (transposed), bias:[3D] f32.
// Q,K:[B,H,S,HD] bf16   Vt:[B,H,HD,S] bf16
__global__ __launch_bounds__(256) void qkv_gemm(const __bf16* __restrict__ xb,
                                                const __bf16* __restrict__ wt,
                                                const float* __restrict__ bias,
                                                __bf16* __restrict__ Q,
                                                __bf16* __restrict__ K,
                                                __bf16* __restrict__ Vt) {
  __shared__ __bf16 ldsA[32 * LDSA_STRIDE];
  const int rowTile = blockIdx.x & 255;                   // 256 x 32-row tiles
  const int colGrp  = blockIdx.x >> 8;                    // 0..5
  const int row0 = rowTile * 32;
  const int w    = threadIdx.x >> 5;
  const int lane = threadIdx.x & 31;
  const int col0 = colGrp * 512 + w * 64;
  const int m = lane & 15, h = lane >> 4;

  stage_a_tile32(xb + (size_t)row0 * DMODEL, ldsA, threadIdx.x);
  __syncthreads();

  const __bf16* ap0 = ldsA + m * LDSA_STRIDE + h * 8;
  const __bf16* ap1 = ap0 + 16 * LDSA_STRIDE;
  const __bf16* bp  = wt + (size_t)(col0 + m) * DMODEL + h * 8;

  v8f acc[8] = {v8f{}, v8f{}, v8f{}, v8f{}, v8f{}, v8f{}, v8f{}, v8f{}};
  mm_k1024_2x4(acc, ap0, ap1, bp);

  // Epilogue: col0 is 64-aligned -> one matrix (q/k/v) and one head per wave.
  // D-tile layout: row = row0 + rr*16 + 8h + r, col = lane&15.
  const int mat  = col0 >> 10;
  const int head = (col0 & 1023) >> 6;
#pragma unroll
  for (int rr = 0; rr < 2; ++rr) {
#pragma unroll
    for (int j = 0; j < 4; ++j) {
      const int c  = col0 + j * 16 + m;
      const float bv = bias[c];
      const int hd = c & 63;
#pragma unroll
      for (int r = 0; r < 8; ++r) {
        const int row = row0 + rr * 16 + h * 8 + r;
        const int b = row >> 11, s = row & 2047;
        const float val = acc[rr * 4 + j][r] + bv;
        const size_t bhi = (size_t)(b * NHEAD + head);
        if (mat == 0)      Q[(bhi * SEQ + s) * HD + hd] = (__bf16)val;
        else if (mat == 1) K[(bhi * SEQ + s) * HD + hd] = (__bf16)val;
        else               Vt[(bhi * HD + hd) * SEQ + s] = (__bf16)val;
      }
    }
  }
}

// ----------------------- flash attention (per b,h,16-q tile) ----------------
// Computes S^T = K * Q^T so the softmax reduction is lane-local and the
// probability A-fragment for P*V is a pure in-lane repack (no LDS transpose).
__global__ __launch_bounds__(256) void attn_fwd(const __bf16* __restrict__ Q,
                                                const __bf16* __restrict__ Kb,
                                                const __bf16* __restrict__ Vt,
                                                __bf16* __restrict__ aout) {
  const int wave = blockIdx.x * 8 + (threadIdx.x >> 5);
  const int lane = threadIdx.x & 31;
  const int bh = wave >> 7;                 // S/16 = 128 query tiles
  const int qt = wave & 127;
  const int b = bh >> 4, head = bh & 15;
  const int q0 = qt * 16;
  const int m = lane & 15, h = lane >> 4;

  const size_t base = (size_t)bh * SEQ * HD;
  const __bf16* qp = Q + base;              // [S][64]
  const __bf16* kp = Kb + base;             // [S][64]
  const __bf16* vp = Vt + base;             // [64][S]

  const __bf16* qrow = qp + (size_t)(q0 + m) * HD + h * 8;
  v16bf bq0 = load_frag_b(qrow);
  v16bf bq1 = load_frag_b(qrow + 32);

  v8f o[4] = {v8f{}, v8f{}, v8f{}, v8f{}};
  float mrow = -1e30f, lrow = 0.f;          // per lane: stats for query lane&15

  const int nkb = (q0 + 47) >> 5;           // causal: # of 32-key blocks
  for (int kbk = 0; kbk < nkb; ++kbk) {
    const int key0 = kbk * 32;

    // issue V-fragment loads early; softmax VALU below hides their latency
    v16bf bv[4];
#pragma unroll
    for (int j = 0; j < 4; ++j)
      bv[j] = load_frag_b(vp + (size_t)(j * 16 + m) * SEQ + key0 + h * 8);

    // prefetch next key block's K rows (global_prefetch_b8)
    if (kbk + 1 < nkb) {
      __builtin_prefetch(kp + (size_t)(key0 + 32 + m) * HD, 0, 0);
      __builtin_prefetch(kp + (size_t)(key0 + 48 + m) * HD, 0, 0);
    }

    float p0[8], p1[8];
    float tmax = -1e30f;
#pragma unroll
    for (int t = 0; t < 2; ++t) {
      const int kb16 = key0 + t * 16;
      float* pr = (t == 0) ? p0 : p1;
      if (kb16 <= q0 + 15) {                // wave-uniform skip of masked tile
        const __bf16* krow = kp + (size_t)(kb16 + m) * HD + h * 8;
        v16bf ak0 = load_frag_a(krow);
        v16bf ak1 = load_frag_a(krow + 32);
        v8f st = {};
        st = __builtin_amdgcn_wmma_f32_16x16x32_bf16(false, ak0, false, bq0,
                                                     (short)0, st, false, false);
        st = __builtin_amdgcn_wmma_f32_16x16x32_bf16(false, ak1, false, bq1,
                                                     (short)0, st, false, false);
        // S^T tile: row = key kb16 + 8h + r, col = query q0 + (lane&15)
#pragma unroll
        for (int r = 0; r < 8; ++r) {
          const int key = kb16 + h * 8 + r;
          float s = st[r] * 0.125f;         // 1/sqrt(64)
          if (key > q0 + m) s = -1e30f;     // causal mask
          pr[r] = s;
          tmax = fmaxf(tmax, s);
        }
      } else {
#pragma unroll
        for (int r = 0; r < 8; ++r) pr[r] = -1e30f;
      }
    }

    // online softmax; merge the two lane halves (keys split across h)
    tmax = fmaxf(tmax, __shfl_xor(tmax, 16, 32));
    const float mnew = fmaxf(mrow, tmax);
    const float corr = __expf(mrow - mnew);
    mrow = mnew;

    float ls = 0.f;
#pragma unroll
    for (int r = 0; r < 8; ++r) {
      p0[r] = __expf(p0[r] - mnew);
      p1[r] = __expf(p1[r] - mnew);
      ls += p0[r] + p1[r];
    }
    ls += __shfl_xor(ls, 16, 32);
    lrow = lrow * corr + ls;

    // rescale O: O rows are queries 8h+r -> fetch that query's corr by shuffle
#pragma unroll
    for (int r = 0; r < 8; ++r) {
      const float c2 = __shfl(corr, h * 8 + r, 32);
      o[0][r] *= c2; o[1][r] *= c2; o[2][r] *= c2; o[3][r] *= c2;
    }

    // P A-fragment (16 queries x 32 keys): in-lane repack of p0/p1
    v16bf aP;
#pragma unroll
    for (int i = 0; i < 8; ++i) {
      aP[i] = (__bf16)p0[i];
      aP[8 + i] = (__bf16)p1[i];
    }

    // O += P * V (4 hd tiles), V fragments already in flight
#pragma unroll
    for (int j = 0; j < 4; ++j)
      o[j] = __builtin_amdgcn_wmma_f32_16x16x32_bf16(false, aP, false, bv[j],
                                                     (short)0, o[j], false, false);
  }

  // finalize: divide by per-query sum, store bf16 [B,S,D]
#pragma unroll
  for (int r = 0; r < 8; ++r) {
    const float li = __shfl(lrow, h * 8 + r, 32);
    const float inv = 1.f / li;             // li >= 1 (self key always present)
    const int s = q0 + h * 8 + r;
    __bf16* orow = aout + ((size_t)b * SEQ + s) * DMODEL + head * HD + m;
    orow[0]  = (__bf16)(o[0][r] * inv);
    orow[16] = (__bf16)(o[1][r] * inv);
    orow[32] = (__bf16)(o[2][r] * inv);
    orow[48] = (__bf16)(o[3][r] * inv);
  }
}

// ---------------------------- output projection -----------------------------
// Block: 32 rows x 512 cols; A tile via async LDS stage; out fp32 [BS][D].
__global__ __launch_bounds__(256) void proj_gemm(const __bf16* __restrict__ A,
                                                 const __bf16* __restrict__ Wt,
                                                 const float* __restrict__ bias,
                                                 float* __restrict__ out) {
  __shared__ __bf16 ldsA[32 * LDSA_STRIDE];
  const int rowTile = blockIdx.x & 255;
  const int colGrp  = blockIdx.x >> 8;                    // 0..1
  const int row0 = rowTile * 32;
  const int w    = threadIdx.x >> 5;
  const int lane = threadIdx.x & 31;
  const int col0 = colGrp * 512 + w * 64;
  const int m = lane & 15, h = lane >> 4;

  stage_a_tile32(A + (size_t)row0 * DMODEL, ldsA, threadIdx.x);
  __syncthreads();

  const __bf16* ap0 = ldsA + m * LDSA_STRIDE + h * 8;
  const __bf16* ap1 = ap0 + 16 * LDSA_STRIDE;
  const __bf16* bp  = Wt + (size_t)(col0 + m) * DMODEL + h * 8;

  v8f acc[8] = {v8f{}, v8f{}, v8f{}, v8f{}, v8f{}, v8f{}, v8f{}, v8f{}};
  mm_k1024_2x4(acc, ap0, ap1, bp);

#pragma unroll
  for (int rr = 0; rr < 2; ++rr) {
#pragma unroll
    for (int j = 0; j < 4; ++j) {
      const int c = col0 + j * 16 + m;
      const float bv = bias[c];
#pragma unroll
      for (int r = 0; r < 8; ++r) {
        const int row = row0 + rr * 16 + h * 8 + r;
        out[(size_t)row * DMODEL + c] = acc[rr * 4 + j][r] + bv;
      }
    }
  }
}

// ------------------------------- launcher -----------------------------------
extern "C" void kernel_launch(void* const* d_in, const int* in_sizes, int n_in,
                              void* d_out, int out_size, void* d_ws, size_t ws_size,
                              hipStream_t stream) {
  (void)in_sizes; (void)n_in; (void)out_size; (void)ws_size;
  const float* x        = (const float*)d_in[0];
  // d_in[1] = mask (bool) — causal structure known statically, unused.
  const float* c_attn_w = (const float*)d_in[2];
  const float* c_attn_b = (const float*)d_in[3];
  const float* c_proj_w = (const float*)d_in[4];
  const float* c_proj_b = (const float*)d_in[5];
  float* out = (float*)d_out;

  // workspace layout (bytes), total 92,274,688 (~88 MB)
  char* ws = (char*)d_ws;
  __bf16* xb    = (__bf16*)(ws + 0);            // 16 MB  [BS][D]
  __bf16* wqkvt = (__bf16*)(ws + 16777216);     //  6 MB  [3D][D]
  __bf16* wprjt = (__bf16*)(ws + 23068672);     //  2 MB  [D][D]
  __bf16* Q     = (__bf16*)(ws + 25165824);     // 16 MB  [B,H,S,HD]
  __bf16* K     = (__bf16*)(ws + 41943040);     // 16 MB  [B,H,S,HD]
  __bf16* Vt    = (__bf16*)(ws + 58720256);     // 16 MB  [B,H,HD,S]
  __bf16* aout  = (__bf16*)(ws + 75497472);     // 16 MB  [BS][D]

  // 1) precision staging: fp32 -> bf16 (weights transposed to [N][K])
  cvt_f32_bf16<<<BS * DMODEL / 256, 256, 0, stream>>>(x, xb, BS * DMODEL);
  tcvt_f32_bf16<<<DMODEL * 3 * DMODEL / 256, 256, 0, stream>>>(c_attn_w, wqkvt,
                                                               DMODEL, 3 * DMODEL);
  tcvt_f32_bf16<<<DMODEL * DMODEL / 256, 256, 0, stream>>>(c_proj_w, wprjt,
                                                           DMODEL, DMODEL);
  // 2) QKV projection: 256 x 32-row tiles, 6 col groups (512 cols/block)
  qkv_gemm<<<1536, 256, 0, stream>>>(xb, wqkvt, c_attn_b, Q, K, Vt);
  // 3) attention: 64 (b,h) x 128 query tiles = 8192 waves
  attn_fwd<<<1024, 256, 0, stream>>>(Q, K, Vt, aout);
  // 4) output projection: 256 x 32-row tiles, 2 col groups
  proj_gemm<<<512, 256, 0, stream>>>(aout, wprjt, c_proj_b, out);
}
